// VSMLRNN_24807731101740
// MI455X (gfx1250) — matrix operand order repaired and checked by reading, not compiled, not run on hardware
//
#include <hip/hip_runtime.h>
#include <hip/hip_bf16.h>

// ---------------------------------------------------------------------------
// VSML RNN forward tick on gfx1250 (MI455X).
//   - gates GEMM (K=32 exactly) via v_wmma_f32_16x16x32_f16, bias in C operand
//   - fwd-message GEMM (K=16 padded to 32, Wf padded to 16 cols) via one WMMA
//   - h C-layout -> A-layout transpose staged through per-wave LDS (stride 20)
//   - LN over MSG=8 via shfl_xor tree; I-reduction via regs + f32 global atomics
//   - bwd path (Wb/bb/lnb) is dead code in the reference -> skipped entirely
// Roofline: ~103 MB mandatory h0/c0 traffic -> ~4.4us @ 23.3 TB/s; WMMA makes
// the 6.6 GFLOP of GEMM negligible, so the kernel is HBM-streaming bound.
// ---------------------------------------------------------------------------

typedef __attribute__((ext_vector_type(16))) _Float16 v16h;
typedef __attribute__((ext_vector_type(8)))  float    v8f;

#define WAVES   8      // 256-thread block = 8 wave32
#define LDS_STR 20     // padded row stride (floats): 16B-aligned rows, no bank conflicts

__device__ __forceinline__ float fast_tanh(float x) {
#if __has_builtin(__builtin_amdgcn_tanhf)
    return __builtin_amdgcn_tanhf(x);           // v_tanh_f32 (gfx1250 TRANS op)
#else
    float e = __builtin_amdgcn_exp2f(x * 2.8853900817779268f);   // e^{2x}
    return 1.0f - 2.0f / (e + 1.0f);
#endif
}
__device__ __forceinline__ float fast_sigm(float x) {
    return 0.5f * fast_tanh(0.5f * x) + 0.5f;
}

// One layer of the VSML net: grid = B*O blocks, block = 256 threads.
// Each wave processes 16-cell tiles along I for its (b,o), runs 2 micro-ticks
// with h,c resident, LayerNorms the final fwd message and atomically reduces
// over I into facc_out[(b*O+o)*8 + m].
__global__ __launch_bounds__(256) void vsml_layer_kernel(
    const float* __restrict__ fwd, int fwd_enc,   // enc: (B,I) scalar in ch0, else (B,I,8)
    const float* __restrict__ bwd, int bwd_enc,   // enc: (B,O) scalar in ch0, else (B,O,8)
    const float* __restrict__ h0,  const float* __restrict__ c0,
    const float* __restrict__ Wl,  const float* __restrict__ bl,   // (32,64),(64)
    const float* __restrict__ Wf,  const float* __restrict__ bf,   // (16,8),(8)
    const float* __restrict__ lnS, const float* __restrict__ lnO,  // (8),(8)
    float* __restrict__ facc_out, int I, int O)
{
    __shared__ __align__(16) float lds[WAVES * 16 * LDS_STR];

    const int tid  = threadIdx.x;
    const int lane = tid & 31;
    const int wav  = tid >> 5;
    const int half = lane >> 4;      // lane half selects K sub-range of A/B
    const int m16  = lane & 15;      // A row (cell) / C column (channel)
    const int bo   = blockIdx.x;
    const int b    = bo / O;
    const int o    = bo - b * O;
    float* myLds = &lds[wav * 16 * LDS_STR];

    // ---- B fragments: W_lstm (32x64) as four 32x16 N-tiles; per-lane column,
    //      K = e + 8*half for e < 8, 16 + 8*half + (e-8) for e >= 8 (matches A)
    v16h Bl0, Bl1, Bl2, Bl3;
    #pragma unroll
    for (int e = 0; e < 16; ++e) {
        const int k = (e < 8) ? (e + 8 * half) : (16 + 8 * half + (e - 8));
        Bl0[e] = (_Float16)Wl[k * 64 +  0 + m16];
        Bl1[e] = (_Float16)Wl[k * 64 + 16 + m16];
        Bl2[e] = (_Float16)Wl[k * 64 + 32 + m16];
        Bl3[e] = (_Float16)Wl[k * 64 + 48 + m16];
    }
    // Wf (16x8) zero-padded to 32x16 (rows 16-31 and cols 8-15 are zero)
    v16h Bf;
    #pragma unroll
    for (int e = 0; e < 16; ++e) {
        float v = 0.0f;
        if (e < 8 && m16 < 8) v = Wf[(e + 8 * half) * 8 + m16];
        Bf[e] = (_Float16)v;
    }
    const float bias0 = bl[ 0 + m16], bias1 = bl[16 + m16];
    const float bias2 = bl[32 + m16], bias3 = bl[48 + m16];
    const float bfv = (m16 < 8) ? bf[m16]  : 0.0f;
    const float lns = (m16 < 8) ? lnS[m16] : 0.0f;
    const float lno = (m16 < 8) ? lnO[m16] : 0.0f;

    // bwd message portion of A (lanes 16-31, elements 0..7) is tile-invariant
    float bwdv[8];
    #pragma unroll
    for (int e = 0; e < 8; ++e) {
        float v = 0.0f;
        if (bwd_enc) { if (e == 0) v = bwd[b * O + o]; }
        else         v = bwd[(b * O + o) * 8 + e];
        bwdv[e] = v;
    }

    float facc = 0.0f;                       // per-lane fwd accumulator (channel m16)
    const int ntiles = I >> 4;

    for (int it = wav; it < ntiles; it += WAVES) {
        const int  i0    = it * 16;
        const long rowA  = (long)(b * I + i0 + m16);       // this lane's A-row cell
        const long cbase = (rowA * O + o) * 16;            // h/c element base

        // prefetch next tile of this wave's streaming state
        __builtin_prefetch(h0 + cbase + (long)16 * WAVES * O * 16, 0, 0);
        __builtin_prefetch(c0 + cbase + (long)16 * WAVES * O * 16, 0, 0);

        // ---- A fragment, elements 0..7: fwd msg (half 0) / bwd msg (half 1)
        v16h a;
        if (half == 0) {
            const int i = i0 + m16;
            if (fwd_enc) {
                a[0] = (_Float16)fwd[b * I + i];
                #pragma unroll
                for (int e = 1; e < 8; ++e) a[e] = (_Float16)0.0f;
            } else {
                const float4 f0 = *(const float4*)&fwd[(long)(b * I + i) * 8];
                const float4 f1 = *(const float4*)&fwd[(long)(b * I + i) * 8 + 4];
                a[0]=(_Float16)f0.x; a[1]=(_Float16)f0.y; a[2]=(_Float16)f0.z; a[3]=(_Float16)f0.w;
                a[4]=(_Float16)f1.x; a[5]=(_Float16)f1.y; a[6]=(_Float16)f1.z; a[7]=(_Float16)f1.w;
            }
        } else {
            #pragma unroll
            for (int e = 0; e < 8; ++e) a[e] = (_Float16)bwdv[e];
        }
        // ---- A elements 8..15: h0[cell][8*half .. 8*half+7]  (K = 16..31)
        {
            const float4 hA = *(const float4*)&h0[cbase + 8 * half];
            const float4 hB = *(const float4*)&h0[cbase + 8 * half + 4];
            a[ 8]=(_Float16)hA.x; a[ 9]=(_Float16)hA.y; a[10]=(_Float16)hA.z; a[11]=(_Float16)hA.w;
            a[12]=(_Float16)hB.x; a[13]=(_Float16)hB.y; a[14]=(_Float16)hB.z; a[15]=(_Float16)hB.w;
        }
        // ---- c in WMMA C-layout: reg r <-> cell i0+r+8*half, channel m16
        float cv[8], hv[8];
        #pragma unroll
        for (int r = 0; r < 8; ++r)
            cv[r] = c0[((long)(b * I + i0 + r + 8 * half) * O + o) * 16 + m16];

        // ==== two micro-ticks ====
        #pragma unroll
        for (int tick = 0; tick < 2; ++tick) {
            v8f g0, g1, g2, g3;
            #pragma unroll
            for (int j = 0; j < 8; ++j) { g0[j]=bias0; g1[j]=bias1; g2[j]=bias2; g3[j]=bias3; }
            g0 = __builtin_amdgcn_wmma_f32_16x16x32_f16(false, a, false, Bl0, (short)0, g0, false, false);
            g1 = __builtin_amdgcn_wmma_f32_16x16x32_f16(false, a, false, Bl1, (short)0, g1, false, false);
            g2 = __builtin_amdgcn_wmma_f32_16x16x32_f16(false, a, false, Bl2, (short)0, g2, false, false);
            g3 = __builtin_amdgcn_wmma_f32_16x16x32_f16(false, a, false, Bl3, (short)0, g3, false, false);

            // LSTM cell update (C-layout elementwise; gate order i,g,f,o)
            #pragma unroll
            for (int r = 0; r < 8; ++r) {
                const float cn = fast_sigm(g2[r] + 1.0f) * cv[r]
                               + fast_sigm(g0[r]) * fast_tanh(g1[r]);
                cv[r] = cn;
                hv[r] = fast_sigm(g3[r]) * fast_tanh(cn);
            }
            // stage h (C-layout) into LDS row-major [cell][s]
            #pragma unroll
            for (int r = 0; r < 8; ++r)
                myLds[(r + 8 * half) * LDS_STR + m16] = hv[r];
            __asm volatile("s_wait_dscnt 0" ::: "memory");

            if (tick == 0) {    // rebuild A h-part for micro-tick 2
                const float* p = &myLds[m16 * LDS_STR + 8 * half];
                const float4 q0 = *(const float4*)p;
                const float4 q1 = *(const float4*)(p + 4);
                a[ 8]=(_Float16)q0.x; a[ 9]=(_Float16)q0.y; a[10]=(_Float16)q0.z; a[11]=(_Float16)q0.w;
                a[12]=(_Float16)q1.x; a[13]=(_Float16)q1.y; a[14]=(_Float16)q1.z; a[15]=(_Float16)q1.w;
            }
        }

        // ==== fwd message: F = h2 @ [Wf | 0] + bf  (K=0..15 valid, rest zero)
        v16h af;
        {
            const float* p = &myLds[m16 * LDS_STR + 8 * half];
            const float4 q0 = *(const float4*)p;
            const float4 q1 = *(const float4*)(p + 4);
            af[0]=(_Float16)q0.x; af[1]=(_Float16)q0.y; af[2]=(_Float16)q0.z; af[3]=(_Float16)q0.w;
            af[4]=(_Float16)q1.x; af[5]=(_Float16)q1.y; af[6]=(_Float16)q1.z; af[7]=(_Float16)q1.w;
            #pragma unroll
            for (int j = 8; j < 16; ++j) af[j] = (_Float16)0.0f;
        }
        v8f F;
        #pragma unroll
        for (int j = 0; j < 8; ++j) F[j] = bfv;
        F = __builtin_amdgcn_wmma_f32_16x16x32_f16(false, af, false, Bf, (short)0, F, false, false);

        // LayerNorm over MSG=8 (xor-tree within each 8-lane group) + cell sum
        #pragma unroll
        for (int r = 0; r < 8; ++r) {
            float y = F[r];
            float s = y, q = y * y;
            s += __shfl_xor(s, 1);  q += __shfl_xor(q, 1);
            s += __shfl_xor(s, 2);  q += __shfl_xor(q, 2);
            s += __shfl_xor(s, 4);  q += __shfl_xor(q, 4);
            const float mean = s * 0.125f;
            const float var  = q * 0.125f - mean * mean;
            const float rs   = __builtin_amdgcn_rsqf(var + 1e-5f);
            facc += (y - mean) * rs * lns + lno;   // lns=lno=0 on pad lanes
        }
    }

    // fold the two lane halves (cells 0-7 + cells 8-15), then reduce over I
    facc += __shfl_xor(facc, 16);
    if (lane < 8)
        atomicAdd(&facc_out[(long)(b * O + o) * 8 + lane], facc);
}

__global__ void vsml_finalize_kernel(const float* __restrict__ acc,
                                     float* __restrict__ out, int n)
{
    const int t = blockIdx.x * blockDim.x + threadIdx.x;
    if (t < n)
        out[t] = 5.0f * fast_tanh(acc[t * 8] * 0.2f);   // channel 0, bounded tanh
}

extern "C" void kernel_launch(void* const* d_in, const int* in_sizes, int n_in,
                              void* d_out, int out_size, void* d_ws, size_t ws_size,
                              hipStream_t stream)
{
    (void)in_sizes; (void)n_in; (void)out_size; (void)ws_size;
    const float* inp  = (const float*)d_in[0];   // (8,784)
    const float* err  = (const float*)d_in[1];   // (8,10)
    const float* bwd0 = (const float*)d_in[2];   // (8,128,8)
    const float* h0_0 = (const float*)d_in[3];   // (8,784,128,16)
    const float* c0_0 = (const float*)d_in[4];
    const float* h0_1 = (const float*)d_in[5];   // (8,128,10,16)
    const float* c0_1 = (const float*)d_in[6];
    const float* Wl   = (const float*)d_in[7];   // (32,64)
    const float* bl   = (const float*)d_in[8];   // (64)
    const float* Wf   = (const float*)d_in[9];   // (16,8)
    const float* bf   = (const float*)d_in[10];  // (8)
    // d_in[11]=Wb, d_in[12]=bb, d_in[15..16]=lnb_* : dead in the reference
    const float* lnfs = (const float*)d_in[13];
    const float* lnfo = (const float*)d_in[14];
    float* out  = (float*)d_out;

    float* acc0 = (float*)d_ws;                  // layer0 fwd_out (8,128,8)
    float* acc1 = acc0 + 8 * 128 * 8;            // layer1 fwd_out (8,10,8)
    hipMemsetAsync(d_ws, 0, (size_t)(8 * 128 * 8 + 8 * 10 * 8) * sizeof(float), stream);

    // layer 0: I=784 (49 tiles/wave-group), O=128, fwd=encode(inp), bwd=bwd_in_0
    vsml_layer_kernel<<<8 * 128, 256, 0, stream>>>(
        inp, 1, bwd0, 0, h0_0, c0_0, Wl, bl, Wf, bf, lnfs, lnfo, acc0, 784, 128);

    // layer 1: I=128, O=10, fwd=layer0 message, bwd=encode(error)
    vsml_layer_kernel<<<8 * 10, 256, 0, stream>>>(
        acc0, 0, err, 1, h0_1, c0_1, Wl, bl, Wf, bf, lnfs, lnfo, acc1, 128, 10);

    vsml_finalize_kernel<<<1, 128, 0, stream>>>(acc1, out, 8 * 10);
}
